// GCNLayer_10153302687994
// MI455X (gfx1250) — matrix operand order repaired and checked by reading, not compiled, not run on hardware
//
#include <hip/hip_runtime.h>

typedef __attribute__((ext_vector_type(16))) _Float16 v16h;
typedef __attribute__((ext_vector_type(8)))  float    v8f;

#define D 64
#define BN_EPS 1e-5f

// deg[i] = 1 (self loop); zero the 128 BN accumulators
__global__ void k_deg_init(float* __restrict__ deg, float* __restrict__ sums, int n) {
    int t = blockIdx.x * blockDim.x + threadIdx.x;
    if (t < 128) sums[t] = 0.0f;
    if (t < n)   deg[t] = 1.0f;
}

// deg[row[e]] += 1
__global__ void k_deg_acc(const int* __restrict__ row, float* __restrict__ deg, int e) {
    int t = blockIdx.x * blockDim.x + threadIdx.x;
    if (t < e) atomicAdd(&deg[row[t]], 1.0f);
}

// agg = self-loop message: x[i]/deg[i]  (norm_self = deg^-0.5 * deg^-0.5)
__global__ void k_agg_init(const float* __restrict__ x, const float* __restrict__ deg,
                           float* __restrict__ agg, int total) {
    int t = blockIdx.x * blockDim.x + threadIdx.x;
    if (t < total) agg[t] = x[t] / deg[t >> 6];
}

// one wave32 per edge: lanes cover 64 features as float2; atomics into agg[row]
__global__ void k_spmm(const int* __restrict__ row, const int* __restrict__ col,
                       const float* __restrict__ x, const float* __restrict__ deg,
                       float* __restrict__ agg, int e) {
    int gid  = blockIdx.x * blockDim.x + threadIdx.x;
    int edge = gid >> 5;
    int lane = gid & 31;
    if (edge >= e) return;
    int r = row[edge];
    int c = col[edge];
    float norm = rsqrtf(deg[r] * deg[c]);          // deg_r^-.5 * deg_c^-.5
    int f = lane << 1;
    float2 v = *(const float2*)(x + c * D + f);
    atomicAdd(agg + r * D + f,     norm * v.x);
    atomicAdd(agg + r * D + f + 1, norm * v.y);
}

// y = relu(agg @ W^T + b) via V_WMMA_F32_16X16X32_F16; fused BN partial sums.
// One wave per 16x16 output tile; block = 8 waves = 32 rows x 64 cols.
__global__ void k_gemm_wmma(const float* __restrict__ agg, const float* __restrict__ W,
                            const float* __restrict__ bias, float* __restrict__ y,
                            float* __restrict__ sums, int n) {
    int lane  = threadIdx.x & 31;
    int w     = threadIdx.x >> 5;
    int mbase = blockIdx.x * 32 + (w >> 2) * 16;
    int obase = (w & 3) * 16;
    if (mbase >= n) return;                        // uniform per wave

    int colx  = lane & 15;                         // tile column (B/C/D layout)
    int hi    = lane >> 4;                         // half-wave selector
    int khalf = hi * 16;                           // A: K sub-offset per half-wave
    int kb    = hi * 16;                           // B: K sub-offset per half-wave

    int mrow = mbase + colx;                       // A: row per lane (lanes&15)
    if (mrow >= n) mrow = n - 1;                   // clamp (keeps EXEC full)

    const float* arow = agg + (long long)mrow * D;
    const float* wrow = W + (obase + colx) * D;    // B[k][o] = W[o][k] -> contiguous in k

    v8f c = {};
#pragma unroll
    for (int kk = 0; kk < D; kk += 32) {
        v16h a, bm;
#pragma unroll
        for (int j = 0; j < 8; ++j) {
            a[j]     = (_Float16)arow[kk + khalf + j];        // VGPR0-3: K=khalf+0..7
            a[8 + j] = (_Float16)arow[kk + khalf + 8 + j];    // VGPR4-7: K=khalf+8..15
        }
#pragma unroll
        for (int j = 0; j < 16; ++j)
            bm[j] = (_Float16)wrow[kk + kb + j];              // K contiguous per lane
        c = __builtin_amdgcn_wmma_f32_16x16x32_f16(
                /*neg_a=*/false, a, /*neg_b=*/false, bm,
                /*c_mod=*/(short)0, c, /*reuse_a=*/false, /*reuse_b=*/false);
    }

    float bv = bias[obase + colx];
    float s = 0.0f, s2 = 0.0f;

    // C/D layout: lane holds column (obase+colx), rows mbase + hi*8 + r, r=0..7
    float* yp = y + (long long)(mbase + hi * 8) * D + obase + colx;

    if (mbase + 16 <= n) {
        // full tile: wave-uniform fast path, straight-line stores (no EXEC churn)
#pragma unroll
        for (int r = 0; r < 8; ++r) {
            float v = c[r] + bv;
            v = v > 0.0f ? v : 0.0f;               // ReLU
            yp[(long long)r * D] = v;
            s  += v;
            s2 += v * v;
        }
    } else {
        // cold tail path (unused when n % 32 == 0)
#pragma unroll
        for (int r = 0; r < 8; ++r) {
            float v = c[r] + bv;
            v = v > 0.0f ? v : 0.0f;
            int m = mbase + hi * 8 + r;
            if (m < n) {
                yp[(long long)r * D] = v;
                s  += v;
                s2 += v * v;
            }
        }
    }

    // fold the two half-waves (same column) and accumulate BN stats
    s  += __shfl_xor(s, 16, 32);
    s2 += __shfl_xor(s2, 16, 32);
    if (lane < 16) {
        atomicAdd(&sums[obase + colx],      s);
        atomicAdd(&sums[64 + obase + colx], s2);
    }
}

// in-place BN finalize on d_out
__global__ void k_bn(float* __restrict__ y, const float* __restrict__ sums,
                     const float* __restrict__ gamma, const float* __restrict__ beta,
                     int n, int total) {
    int t = blockIdx.x * blockDim.x + threadIdx.x;
    if (t >= total) return;
    int o = t & 63;
    float inv_n = 1.0f / (float)n;
    float mean  = sums[o] * inv_n;
    float var   = sums[64 + o] * inv_n - mean * mean;
    float sc    = rsqrtf(var + BN_EPS) * gamma[o];
    y[t] = (y[t] - mean) * sc + beta[o];
}

extern "C" void kernel_launch(void* const* d_in, const int* in_sizes, int n_in,
                              void* d_out, int out_size, void* d_ws, size_t ws_size,
                              hipStream_t stream) {
    const float* x     = (const float*)d_in[0];
    const int*   ei    = (const int*)  d_in[1];
    const float* W     = (const float*)d_in[2];
    const float* b     = (const float*)d_in[3];
    const float* gamma = (const float*)d_in[4];
    const float* beta  = (const float*)d_in[5];

    int n = in_sizes[0] / D;          // 100000
    int e = in_sizes[1] / 2;          // 1200000
    const int* row = ei;
    const int* col = ei + e;

    float* ws   = (float*)d_ws;
    float* deg  = ws;                               // n floats
    float* sums = ws + n;                           // 128 floats (sum | sumsq)
    float* agg  = ws + ((n + 128 + 15) & ~15);      // n*64 floats, 16-float aligned
    float* y    = (float*)d_out;

    const int B = 256;
    int total = n * D;

    k_deg_init<<<(n + B - 1) / B, B, 0, stream>>>(deg, sums, n);
    k_deg_acc <<<(e + B - 1) / B, B, 0, stream>>>(row, deg, e);
    k_agg_init<<<(total + B - 1) / B, B, 0, stream>>>(x, deg, agg, total);

    long long spmm_threads = (long long)e * 32;
    k_spmm<<<(int)((spmm_threads + B - 1) / B), B, 0, stream>>>(row, col, x, deg, agg, e);

    k_gemm_wmma<<<(n + 31) / 32, 256, 0, stream>>>(agg, W, b, y, sums, n);
    k_bn<<<(total + B - 1) / B, B, 0, stream>>>(y, sums, gamma, beta, n, total);
}
